// VQKDVisualTokenizer_30219389894937
// MI455X (gfx1250) — compile-verified, matrix operand order checked by reading, not run on hardware
//
#include <hip/hip_runtime.h>
#include <hip/hip_bf16.h>
#include <math.h>

// Problem constants (from reference): B=64, N=197, D=768, V=8192
#define M_ROWS   12608       // B*N patches
#define D_DIM    768
#define V_DIM    8192
#define LDSP     776         // padded LDS row stride in halves (768+8): breaks bank conflicts
#define NTILES_V (V_DIM/16)  // 512
#define KSTEPS   (D_DIM/32)  // 24
#define WAVES    8
#define COLS_PER_BLOCK 32    // 2 x 16-column WMMA tiles per block

typedef __attribute__((ext_vector_type(16))) _Float16 v16h;
typedef __attribute__((ext_vector_type(8)))  _Float16 v8h;
typedef __attribute__((ext_vector_type(8)))  float    v8f;

union Frag {
    v16h f;
    v8h  h[2];
};

// ---------------------------------------------------------------------------
// Kernel 1: L2-normalize one row of 768 fp32 values, emit f16 row.
// ---------------------------------------------------------------------------
__global__ __launch_bounds__(256) void l2norm_rows_f16(const float* __restrict__ src,
                                                       _Float16* __restrict__ dst) {
    __shared__ float red[256];
    const int row = blockIdx.x;
    const float* s = src + (size_t)row * D_DIM;
    float acc = 0.f;
#pragma unroll
    for (int i = 0; i < 3; ++i) {
        float v = s[threadIdx.x + i * 256];
        acc += v * v;
    }
    red[threadIdx.x] = acc;
    __syncthreads();
    for (int off = 128; off > 0; off >>= 1) {
        if (threadIdx.x < off) red[threadIdx.x] += red[threadIdx.x + off];
        __syncthreads();
    }
    const float inv = 1.0f / fmaxf(sqrtf(red[0]), 1e-12f);
    _Float16* d = dst + (size_t)row * D_DIM;
#pragma unroll
    for (int i = 0; i < 3; ++i) {
        const int j = threadIdx.x + i * 256;
        d[j] = (_Float16)(s[j] * inv);
    }
}

// ---------------------------------------------------------------------------
// Kernel 2: fused cosine-GEMM + argmax over V, via v_wmma_f32_16x16x32_f16.
//   A = codebook tile (M=16 entries x K), B = patch tile (K x N=16 patches).
//   D-tile layout (ISA 7.12.2): lane c (<16) holds rows 0..7 of column c in
//   VGPR 0..7; lane c+16 holds rows 8..15 of column c. So the V-argmax is a
//   per-lane running max + one shfl_xor(16) merge.
//
//   Hot-loop argmax: within one lane the candidate indices are strictly
//   ascending, so strict '>' alone preserves first-max (jnp.argmax) semantics;
//   ternary selects keep it branchless (v_cmp + v_cndmask, no exec juggling).
//   Index tie-breaks are applied only at the cross-lane / cross-wave merges.
// ---------------------------------------------------------------------------
__global__ __launch_bounds__(256) void gemm_argmax(const _Float16* __restrict__ featN,
                                                   const _Float16* __restrict__ cbN,
                                                   int* __restrict__ outIdx) {
    __shared__ _Float16 fe[COLS_PER_BLOCK * LDSP];
    __shared__ float sMax[WAVES * COLS_PER_BLOCK];
    __shared__ int   sIdx[WAVES * COLS_PER_BLOCK];

    const int tid  = threadIdx.x;
    const int wave = tid >> 5;
    const int lane = tid & 31;
    const int n0   = blockIdx.x * COLS_PER_BLOCK;   // first patch column

    // Stage 32 normalized-f16 feature rows into padded LDS (ds_store_b128).
    for (int c = tid; c < COLS_PER_BLOCK * (D_DIM / 8); c += 256) {
        const int r = c / (D_DIM / 8);
        const int k = (c % (D_DIM / 8)) * 8;
        *(v8h*)&fe[r * LDSP + k] = *(const v8h*)&featN[(size_t)(n0 + r) * D_DIM + k];
    }
    __syncthreads();

    const int lrow  = lane & 15;          // A row / B column within tile
    const int khalf = (lane >> 4) * 8;    // 16-bit frag layout: lanes>=16 own K+8,K+24

    float bestV[2] = {-1.0e30f, -1.0e30f};
    int   bestI[2] = {0, 0};

    const int vtPerWave = NTILES_V / WAVES;      // 64 tiles of 16 entries
    const int vtBase    = wave * vtPerWave;

    for (int vt = 0; vt < vtPerWave; ++vt) {
        const int v0 = (vtBase + vt) * 16;
        const _Float16* arow = cbN + (size_t)(v0 + lrow) * D_DIM + khalf;

        v8f acc0 = {};
        v8f acc1 = {};
#pragma unroll 4
        for (int ks = 0; ks < KSTEPS; ++ks) {
            const int k0 = ks * 32;
            Frag a, b0, b1;
            // A: codebook entry row, documented 16-bit 16x32 layout
            a.h[0] = *(const v8h*)(arow + k0);
            a.h[1] = *(const v8h*)(arow + k0 + 16);
            // B: two patch-column tiles from LDS (ds_load_b128, conflict-free)
            b0.h[0] = *(const v8h*)&fe[lrow * LDSP + k0 + khalf];
            b0.h[1] = *(const v8h*)&fe[lrow * LDSP + k0 + khalf + 16];
            b1.h[0] = *(const v8h*)&fe[(16 + lrow) * LDSP + k0 + khalf];
            b1.h[1] = *(const v8h*)&fe[(16 + lrow) * LDSP + k0 + khalf + 16];
            acc0 = __builtin_amdgcn_wmma_f32_16x16x32_f16(false, a.f, false, b0.f,
                                                          (short)0, acc0, false, false);
            acc1 = __builtin_amdgcn_wmma_f32_16x16x32_f16(false, a.f, false, b1.f,
                                                          (short)0, acc1, false, false);
        }

        // Branchless running argmax (indices ascend per lane => '>' keeps first).
        const int ebase = v0 + ((lane >> 4) << 3);
#pragma unroll
        for (int r = 0; r < 8; ++r) {
            const float x0 = acc0[r];
            const bool  c0 = x0 > bestV[0];
            bestV[0] = c0 ? x0 : bestV[0];
            bestI[0] = c0 ? (ebase + r) : bestI[0];

            const float x1 = acc1[r];
            const bool  c1 = x1 > bestV[1];
            bestV[1] = c1 ? x1 : bestV[1];
            bestI[1] = c1 ? (ebase + r) : bestI[1];
        }
    }

    // Merge the two row-halves of each column (lane c <-> lane c+16).
    // Index ranges interleave here, so apply the first-index tie-break.
#pragma unroll
    for (int t = 0; t < 2; ++t) {
        const float ov = __shfl_xor(bestV[t], 16, 32);
        const int   oi = __shfl_xor(bestI[t], 16, 32);
        const bool  c  = (ov > bestV[t]) || (ov == bestV[t] && oi < bestI[t]);
        bestV[t] = c ? ov : bestV[t];
        bestI[t] = c ? oi : bestI[t];
        if (lane < 16) {
            sMax[wave * COLS_PER_BLOCK + t * 16 + lane] = bestV[t];
            sIdx[wave * COLS_PER_BLOCK + t * 16 + lane] = bestI[t];
        }
    }
    __syncthreads();

    // Cross-wave reduce (waves own disjoint ascending V ranges; reduce in wave
    // order with tie-break keeps the first-max index, matching jnp.argmax).
    if (tid < COLS_PER_BLOCK) {
        float bv = sMax[tid];
        int   bi = sIdx[tid];
        for (int w = 1; w < WAVES; ++w) {
            const float v = sMax[w * COLS_PER_BLOCK + tid];
            const int   i = sIdx[w * COLS_PER_BLOCK + tid];
            const bool  c = (v > bv) || (v == bv && i < bi);
            bv = c ? v : bv;
            bi = c ? i : bi;
        }
        outIdx[n0 + tid] = bi;
    }
}

// ---------------------------------------------------------------------------
// Kernel 3: gather winning codebook row (fp32) and L2-normalize -> output.
// ---------------------------------------------------------------------------
__global__ __launch_bounds__(256) void gather_l2norm(const float* __restrict__ cb,
                                                     const int* __restrict__ idx,
                                                     float* __restrict__ out) {
    __shared__ float red[256];
    const int row = blockIdx.x;
    const int v   = idx[row];
    const float* s = cb + (size_t)v * D_DIM;
    float acc = 0.f;
#pragma unroll
    for (int i = 0; i < 3; ++i) {
        const float x = s[threadIdx.x + i * 256];
        acc += x * x;
    }
    red[threadIdx.x] = acc;
    __syncthreads();
    for (int off = 128; off > 0; off >>= 1) {
        if (threadIdx.x < off) red[threadIdx.x] += red[threadIdx.x + off];
        __syncthreads();
    }
    const float inv = 1.0f / fmaxf(sqrtf(red[0]), 1e-12f);
    float* d = out + (size_t)row * D_DIM;
#pragma unroll
    for (int i = 0; i < 3; ++i) {
        const int j = threadIdx.x + i * 256;
        d[j] = s[j] * inv;
    }
}

extern "C" void kernel_launch(void* const* d_in, const int* in_sizes, int n_in,
                              void* d_out, int out_size, void* d_ws, size_t ws_size,
                              hipStream_t stream) {
    const float* feat = (const float*)d_in[0];   // [12608, 768] fp32
    const float* cb   = (const float*)d_in[1];   // [8192, 768] fp32

    // Outputs: quantized_normalized [12608*768] fp32, then indices [12608] int32.
    float* outQ   = (float*)d_out;
    int*   outIdx = (int*)((char*)d_out + (size_t)M_ROWS * D_DIM * sizeof(float));

    // Workspace: normalized f16 copies (19.4 MB + 12.6 MB).
    _Float16* featN = (_Float16*)d_ws;
    _Float16* cbN   = (_Float16*)((char*)d_ws + (size_t)M_ROWS * D_DIM * sizeof(_Float16));

    l2norm_rows_f16<<<M_ROWS, 256, 0, stream>>>(feat, featN);
    l2norm_rows_f16<<<V_DIM, 256, 0, stream>>>(cb, cbN);
    gemm_argmax<<<M_ROWS / COLS_PER_BLOCK, 256, 0, stream>>>(featN, cbN, outIdx);
    gather_l2norm<<<M_ROWS, 256, 0, stream>>>(cb, outIdx, outQ);
}
